// MyModel_87522843558501
// MI455X (gfx1250) — compile-verified, matrix operand order checked by reading, not run on hardware
//
#include <hip/hip_runtime.h>
#include <hip/hip_bf16.h>

typedef __attribute__((ext_vector_type(16))) _Float16 v16h;
typedef __attribute__((ext_vector_type(8)))  _Float16 v8h;
typedef __attribute__((ext_vector_type(8)))  float    v8f;

#define T_STEPS 512
#define EMB_DIM 16
#define HID 32
#define NGATE 128        // 4*HID
#define BTILE 16         // batch rows per wave

// fast sigmoid / tanh on v_exp_f32 + v_rcp_f32 (co-executes with XDL WMMA)
__device__ __forceinline__ float sigf(float x) {
  return __builtin_amdgcn_rcpf(1.0f + __builtin_amdgcn_exp2f(-1.44269504f * x));
}
__device__ __forceinline__ float tanhf_fast(float x) {
  return 2.0f * __builtin_amdgcn_rcpf(1.0f + __builtin_amdgcn_exp2f(-2.88539009f * x)) - 1.0f;
}

// One wave (32 lanes) per 16-row batch tile; whole T-loop with h,c and all
// LSTM weights resident in registers. zT = K^T x^T + R^T h^T via
// v_wmma_f32_16x16x32_f16; C/D layout of h^T is bit-compatible with the next
// step's B-fragment after per-lane f32->f16 packing (no cross-lane traffic).
// The id->emb LDS chain for step t+1 is software-pipelined so its latency
// overlaps step t's recurrent WMMA + gate math.
__global__ __launch_bounds__(32) void lstm_fused_kernel(
    const int* __restrict__ ids,        // [B, 512]
    const float* __restrict__ emb,      // [1000, 16]
    const float* __restrict__ kernel,   // [16, 128]
    const float* __restrict__ rec,      // [32, 128]
    const float* __restrict__ w1,       // [32, 32]
    const float* __restrict__ b1,       // [32]
    const float* __restrict__ w2,       // [32, 1]
    const float* __restrict__ b2,       // [1]
    float* __restrict__ out,            // [B]
    int emb_elems)                      // 16000
{
  __shared__ alignas(16) _Float16 emb_lds[1000 * EMB_DIM];   // 32000 B, f16 table
  __shared__ unsigned short ids_lds[T_STEPS * BTILE];        // [t][b] u16, 16 KB
  __shared__ float hbuf[HID * BTILE];                        // epilogue staging

  const int lane = threadIdx.x;        // wave32
  const int b0   = blockIdx.x * BTILE;
  const int col  = lane & 15;          // batch column / weight column within tile
  const bool hi  = lane >= 16;         // lane half selects K sub-range
  const int eoff = hi ? 8 : 0;         // emb element offset for this lane half

  // ---- stage embedding table as f16 in LDS (read many times per step) ----
  for (int i = lane; i < emb_elems; i += 32)
    emb_lds[i] = (_Float16)emb[i];

  // ---- stage this tile's ids as u16 in [t][b] layout (conflict-free reads) ----
  for (int i = lane; i < BTILE * T_STEPS; i += 32) {
    int b = i >> 9;            // /512
    int t = i & (T_STEPS - 1);
    ids_lds[t * BTILE + b] = (unsigned short)ids[(b0 + b) * T_STEPS + t];
  }

  // ---- build A-matrix fragments: K^T tiles (K padded 16->32) and R^T tiles ----
  // A layout (16-bit, 16x32): lanes 0-15 row M=lane hold K 0-7 / 16-23,
  // lanes 16-31 hold K 8-15 / 24-31 (two halves per VGPR).
  v16h a_k[8], a_r[8];
#pragma unroll
  for (int tile = 0; tile < 8; ++tile) {
    int wcol = tile * 16 + col;        // column of [16|32 x 128] weight
    v16h ak = {};
    v16h ar;
#pragma unroll
    for (int i = 0; i < 8; ++i) {
      int e = eoff + i;
      ak[i] = (_Float16)kernel[e * NGATE + wcol];            // K rows 0..15 (16..31 pad 0)
      ar[i] = (_Float16)rec[e * NGATE + wcol];               // R rows 0..15
      ar[8 + i] = (_Float16)rec[(16 + e) * NGATE + wcol];    // R rows 16..31
    }
    a_k[tile] = ak;
    a_r[tile] = ar;
  }

  __syncthreads();

  // ---- recurrent state: zT acc layout => c/h tiles are elementwise v8f ----
  v8f c0 = {}, c1 = {};     // cell units 0-15 / 16-31  (lane = batch col)
  v8f h0 = {}, h1 = {};     // hidden, same layout
  v16h bh = {};             // h^T as B-fragment (all-zero at t=0)
  const v8f zf = {};

  // prologue of the software pipeline: fetch x fragment for t=0
  int id0 = (int)ids_lds[col];
  v8h xv = *reinterpret_cast<const v8h*>(&emb_lds[id0 * EMB_DIM + eoff]);

#pragma unroll 1
  for (int t = 0; t < T_STEPS; ++t) {
    // x^T B-fragment: lane<16 -> emb[id][0..7], lane>=16 -> emb[id][8..15]; K 16..31 pad 0
    v16h bx = {};
#pragma unroll
    for (int i = 0; i < 8; ++i) bx[i] = xv[i];

    // ---- prefetch id + emb row for t+1: latency hides under this step ----
    int idn = (int)ids_lds[((t + 1) & (T_STEPS - 1)) * BTILE + col];

    // gates i,f,g,o: even tiles -> units 0-15, odd tiles -> units 16-31
    v8f zi0, zi1, zf0, zf1, zg0, zg1, zo0, zo1;
    zi0 = __builtin_amdgcn_wmma_f32_16x16x32_f16(false, a_k[0], false, bx, (short)0, zf,  false, false);
    zf0 = __builtin_amdgcn_wmma_f32_16x16x32_f16(false, a_k[2], false, bx, (short)0, zf,  false, false);
    zg0 = __builtin_amdgcn_wmma_f32_16x16x32_f16(false, a_k[4], false, bx, (short)0, zf,  false, false);
    zo0 = __builtin_amdgcn_wmma_f32_16x16x32_f16(false, a_k[6], false, bx, (short)0, zf,  false, false);
    zi1 = __builtin_amdgcn_wmma_f32_16x16x32_f16(false, a_k[1], false, bx, (short)0, zf,  false, false);
    zf1 = __builtin_amdgcn_wmma_f32_16x16x32_f16(false, a_k[3], false, bx, (short)0, zf,  false, false);
    zg1 = __builtin_amdgcn_wmma_f32_16x16x32_f16(false, a_k[5], false, bx, (short)0, zf,  false, false);
    zo1 = __builtin_amdgcn_wmma_f32_16x16x32_f16(false, a_k[7], false, bx, (short)0, zf,  false, false);

    // second half of the prefetch (depends only on idn)
    v8h xvn = *reinterpret_cast<const v8h*>(&emb_lds[idn * EMB_DIM + eoff]);

    // recurrent WMMAs: the true serial dependence (bh from previous step)
    zi0 = __builtin_amdgcn_wmma_f32_16x16x32_f16(false, a_r[0], false, bh, (short)0, zi0, false, false);
    zf0 = __builtin_amdgcn_wmma_f32_16x16x32_f16(false, a_r[2], false, bh, (short)0, zf0, false, false);
    zg0 = __builtin_amdgcn_wmma_f32_16x16x32_f16(false, a_r[4], false, bh, (short)0, zg0, false, false);
    zo0 = __builtin_amdgcn_wmma_f32_16x16x32_f16(false, a_r[6], false, bh, (short)0, zo0, false, false);

    // update units 0-15 while the odd-tile WMMAs can overlap
#pragma unroll
    for (int r = 0; r < 8; ++r) {
      float cv = sigf(zf0[r]) * c0[r] + sigf(zi0[r]) * tanhf_fast(zg0[r]);
      c0[r] = cv;
      h0[r] = sigf(zo0[r]) * tanhf_fast(cv);
    }

    zi1 = __builtin_amdgcn_wmma_f32_16x16x32_f16(false, a_r[1], false, bh, (short)0, zi1, false, false);
    zf1 = __builtin_amdgcn_wmma_f32_16x16x32_f16(false, a_r[3], false, bh, (short)0, zf1, false, false);
    zg1 = __builtin_amdgcn_wmma_f32_16x16x32_f16(false, a_r[5], false, bh, (short)0, zg1, false, false);
    zo1 = __builtin_amdgcn_wmma_f32_16x16x32_f16(false, a_r[7], false, bh, (short)0, zo1, false, false);

#pragma unroll
    for (int r = 0; r < 8; ++r) {
      float cv = sigf(zf1[r]) * c1[r] + sigf(zi1[r]) * tanhf_fast(zg1[r]);
      c1[r] = cv;
      h1[r] = sigf(zo1[r]) * tanhf_fast(cv);
    }

    // repack h^T into next step's B-fragment: purely lane-local (layout identity)
    v16h bhn;
#pragma unroll
    for (int i = 0; i < 8; ++i) {
      bhn[i]     = (_Float16)h0[i];
      bhn[8 + i] = (_Float16)h1[i];
    }
    bh = bhn;
    xv = xvn;
  }

  // ---- epilogue: stage h^T [unit][batch] to LDS, then per-batch MLP ----
#pragma unroll
  for (int r = 0; r < 8; ++r) {
    int u = eoff + r;
    hbuf[u * BTILE + col]        = h0[r];
    hbuf[(16 + u) * BTILE + col] = h1[r];
  }
  __syncthreads();

  if (lane < BTILE) {
    float hv[HID];
#pragma unroll
    for (int k = 0; k < HID; ++k) hv[k] = hbuf[k * BTILE + lane];
    float acc = b2[0];
#pragma unroll 4
    for (int j = 0; j < 32; ++j) {
      float s = b1[j];
#pragma unroll
      for (int k = 0; k < HID; ++k) s = fmaf(hv[k], w1[k * 32 + j], s);
      acc = fmaf(w2[j], fmaxf(s, 0.0f), acc);
    }
    out[b0 + lane] = acc;
  }
}

extern "C" void kernel_launch(void* const* d_in, const int* in_sizes, int n_in,
                              void* d_out, int out_size, void* d_ws, size_t ws_size,
                              hipStream_t stream) {
  const int*   ids    = (const int*)d_in[0];
  const float* emb    = (const float*)d_in[1];
  const float* kernel = (const float*)d_in[2];
  const float* rec    = (const float*)d_in[3];
  const float* w1     = (const float*)d_in[4];
  const float* b1     = (const float*)d_in[5];
  const float* w2     = (const float*)d_in[6];
  const float* b2     = (const float*)d_in[7];
  float* out = (float*)d_out;

  int B = in_sizes[0] / T_STEPS;          // 4096
  int grid = B / BTILE;                   // 256 waves, one per batch tile

  lstm_fused_kernel<<<grid, 32, 0, stream>>>(ids, emb, kernel, rec, w1, b1, w2, b2,
                                             out, in_sizes[1]);
}
